// EdgeClassifier_66632122630724
// MI455X (gfx1250) — compile-verified, matrix operand order checked by reading, not run on hardware
//
#include <hip/hip_runtime.h>
#include <hip/hip_bf16.h>
#include <math.h>

// ---------------- problem constants ----------------
#define NN      1024          // nodes
#define DD      128           // embedding dim
#define TWO_D   256           // feat dim (2D)
#define HH      512           // hidden
#define HH2     256           // hidden/2
#define EE      523776        // N*(N-1)/2 edges
#define EPSLN   1e-5f

typedef __attribute__((ext_vector_type(16))) _Float16 v16h;
typedef __attribute__((ext_vector_type(8)))  _Float16 v8h;
typedef __attribute__((ext_vector_type(8)))  float    v8f;

// TDM descriptor SGPR-group types (probe-verified shapes)
typedef __attribute__((ext_vector_type(4))) unsigned int tdm_g0_t;
typedef __attribute__((ext_vector_type(8))) int          tdm_g1_t;
typedef __attribute__((ext_vector_type(4))) int          tdm_g2_t;

// ---------------- workspace layout (f16 staging) ----------------
// embH : [1024][128]  f16   (256 KB)
// W1t  : [512][256]   f16   N-major transpose of W1[256][512]  (256 KB)
// W2t  : [256][512]   f16   N-major transpose of W2[512][256]  (256 KB)
#define EMBH_ELEMS  (NN * DD)          // 131072
#define W1T_ELEMS   (HH * TWO_D)       // 131072
#define W2T_ELEMS   (HH2 * HH)         // 131072

// ---------------- LDS layout (dynamic, per 256-thread block) ----------------
// Bstage   : 2 x 16384 B double-buffered TDM weight panels      = 32768
// feats/h2 : 8 waves * 16*256 f16 (reused as h2 after GEMM2)    = 65536
// h1       : 8 waves * 16*512 f16                               = 131072
// stats    : 8 waves * 32 f32 (mean, rstd per row)              = 1024
// consts   : b1(512)+b2(256)+gamma(512)+beta(512)+W3(256) f32   = 8192
#define LDS_BSTAGE_OFF  0
#define LDS_FEATS_OFF   32768
#define LDS_H1_OFF      98304
#define LDS_STATS_OFF   229376
#define LDS_CONST_OFF   230400
#define LDS_TOTAL       238592          // 233 KB: needs CDNA5's 320 KB WGP LDS

#define QW_PANEL1   1024   // GEMM1 B panel: 16n x 256k f16 = 8 KB  = 1024 x 8B
#define QW_PANEL2   2048   // GEMM2 B panel: 16n x 512k f16 = 16 KB = 2048 x 8B

__device__ __forceinline__ int triu_start(int i) {
    // # of edges whose first index < i  ==  i*(2N-1-i)/2
    return (int)(((long long)i * (2 * NN - 1 - i)) >> 1);
}

// ---- Tensor Data Mover: linear copy of nqw 8-byte units global -> LDS ----
// D# per CDNA5 ISA ch.8: group0 = {count | lds_addr | global_addr | type=2},
// group1 = {data_size=8B, tensor_dim0=nqw, tensor_dim1=1, tile_dim0=nqw,
//           tile_dim1=1, tensor_dim0_stride=nqw}. Groups 2/3 unused (<=2D).
__device__ __forceinline__ void tdm_load_lin(unsigned int lds_byte,
                                             unsigned long long gaddr,
                                             unsigned int nqw) {
    tdm_g0_t g0;
    g0[0] = 1u;                                              // count=1 (valid)
    g0[1] = lds_byte;                                        // lds_addr
    g0[2] = (unsigned int)gaddr;                             // global_addr[31:0]
    g0[3] = (unsigned int)((gaddr >> 32) & 0x01FFFFFFu)      // global_addr[56:32]
          | (2u << 30);                                      // type=2 ("image")
    tdm_g1_t g1;
    g1[0] = (int)(3u << 16);                                 // data_size=3 (8B), mask=0
    g1[1] = (int)((nqw & 0xFFFFu) << 16);                    // tensor_dim0[15:0]
    g1[2] = (int)(((nqw >> 16) & 0xFFFFu) | (1u << 16));     // tensor_dim0[31:16], tensor_dim1=1
    g1[3] = (int)((nqw & 0xFFFFu) << 16);                    // tile_dim0 = nqw
    g1[4] = 1;                                               // tile_dim1=1, tile_dim2=0
    g1[5] = (int)nqw;                                        // tensor_dim0_stride[31:0]
    g1[6] = 0;                                               // stride0 hi / stride1 lo
    g1[7] = 0;
    tdm_g2_t gz = {0, 0, 0, 0};
#if defined(__clang_major__) && (__clang_major__ >= 23)
    tdm_g1_t gz8 = {0, 0, 0, 0, 0, 0, 0, 0};
    __builtin_amdgcn_tensor_load_to_lds(g0, g1, gz, gz, gz8, 0);
#else
    __builtin_amdgcn_tensor_load_to_lds(g0, g1, gz, gz, 0);
#endif
}

// ================= pack: f32 -> f16, transpose weights to N-major =============
__global__ void __launch_bounds__(256)
edge_pack_kernel(const float* __restrict__ emb,
                 const float* __restrict__ W1,
                 const float* __restrict__ W2,
                 _Float16* __restrict__ embH,
                 _Float16* __restrict__ W1t,
                 _Float16* __restrict__ W2t) {
    int idx = blockIdx.x * blockDim.x + threadIdx.x;
    if (idx < EMBH_ELEMS) {
        embH[idx] = (_Float16)emb[idx];
    } else if (idx < EMBH_ELEMS + W1T_ELEMS) {
        int t = idx - EMBH_ELEMS;
        int n = t >> 8;            // 0..511
        int k = t & 255;           // 0..255
        W1t[t] = (_Float16)W1[k * HH + n];
    } else if (idx < EMBH_ELEMS + W1T_ELEMS + W2T_ELEMS) {
        int t = idx - EMBH_ELEMS - W1T_ELEMS;
        int n = t >> 9;            // 0..255
        int k = t & 511;           // 0..511
        W2t[t] = (_Float16)W2[k * HH2 + n];
    }
}

// ================= main fused edge-MLP kernel ================================
__global__ void __launch_bounds__(256)
edge_mlp_kernel(const _Float16* __restrict__ embH,
                const _Float16* __restrict__ W1t,
                const _Float16* __restrict__ W2t,
                const float* __restrict__ b1,
                const float* __restrict__ gamma,
                const float* __restrict__ beta,
                const float* __restrict__ b2,
                const float* __restrict__ W3,
                const float* __restrict__ b3,
                float* __restrict__ out) {   // out: [E logits][2E pairs]
    extern __shared__ char smem[];

    const int tid  = threadIdx.x;
    const int lane = tid & 31;
    const int wv   = tid >> 5;           // wave 0..7
    const int m    = lane & 15;          // row within 16-edge tile
    const int sub  = lane >> 4;          // half-wave group

    _Float16* bstage = (_Float16*)(smem + LDS_BSTAGE_OFF);       // 2 x 16 KB
    _Float16* featsW = (_Float16*)(smem + LDS_FEATS_OFF) + wv * 16 * TWO_D;
    _Float16* h1W    = (_Float16*)(smem + LDS_H1_OFF)    + wv * 16 * HH;
    float*    statsW = (float*)(smem + LDS_STATS_OFF)    + wv * 32;
    float*    cB1    = (float*)(smem + LDS_CONST_OFF);
    float*    cB2    = cB1 + HH;
    float*    cGamma = cB2 + HH2;
    float*    cBeta  = cGamma + HH;
    float*    cW3    = cBeta + HH;

    const unsigned int bstage_lds0 = (unsigned int)(uintptr_t)(smem + LDS_BSTAGE_OFF);
    const unsigned int bstage_lds1 = bstage_lds0 + 16384u;

    // ---- stage broadcast constants into LDS once per block ----
    for (int i = tid; i < HH;  i += 256) cB1[i]    = b1[i];
    for (int i = tid; i < HH2; i += 256) cB2[i]    = b2[i];
    for (int i = tid; i < HH;  i += 256) cGamma[i] = gamma[i];
    for (int i = tid; i < HH;  i += 256) cBeta[i]  = beta[i];
    for (int i = tid; i < HH2; i += 256) cW3[i]    = W3[i];
    __syncthreads();

    // ---- TDM prologue: kick off GEMM1 weight panel 0 while we build feats ----
    if (tid == 0) {
        tdm_load_lin(bstage_lds0, (unsigned long long)(uintptr_t)W1t, QW_PANEL1);
    }

    const int tileBase = (blockIdx.x * 8 + wv) * 16;   // first edge of this wave's tile
    const int e = tileBase + m;                        // each lane tracks its row's edge

    // ---- invert upper-triangular linear index -> (i, j) ----
    double disc = 2047.0 * 2047.0 - 8.0 * (double)e;
    int iIdx = (int)((2047.0 - sqrt(disc)) * 0.5);
    if (iIdx < 0) iIdx = 0;
    if (iIdx > NN - 2) iIdx = NN - 2;
    while (iIdx > 0 && triu_start(iIdx) > e) --iIdx;
    while (triu_start(iIdx + 1) <= e) ++iIdx;
    const int jIdx = iIdx + 1 + (e - triu_start(iIdx));

    // ---- build feats tile [16 x 256] f16 in LDS ----
    // lanes 0..15 write diff half (cols 0..127), lanes 16..31 product half (128..255)
    {
        const _Float16* ei = embH + iIdx * DD;
        const _Float16* ej = embH + jIdx * DD;
        _Float16* dst = featsW + m * TWO_D + sub * DD;
        if (sub == 0) {
            #pragma unroll
            for (int k8 = 0; k8 < DD / 8; ++k8) {
                v8h a = *(const v8h*)(ei + k8 * 8);
                v8h b = *(const v8h*)(ej + k8 * 8);
                *(v8h*)(dst + k8 * 8) = a - b;
            }
        } else {
            #pragma unroll
            for (int k8 = 0; k8 < DD / 8; ++k8) {
                v8h a = *(const v8h*)(ei + k8 * 8);
                v8h b = *(const v8h*)(ej + k8 * 8);
                *(v8h*)(dst + k8 * 8) = a * b;
            }
        }
    }

    // ---- GEMM1: h1 = relu(feats @ W1 + b1), fused LN-stat accumulation ----
    // B panels TDM-staged in LDS, double-buffered, shared by all 8 waves.
    float sum[8], sumsq[8];
    #pragma unroll
    for (int r = 0; r < 8; ++r) { sum[r] = 0.f; sumsq[r] = 0.f; }

    const int nl = m;  // C-fragment column for this lane
    #pragma unroll 1
    for (int nt = 0; nt < HH / 16; ++nt) {           // 32 n-tiles
        if (tid == 0) {
            if (nt + 1 < HH / 16) {
                tdm_load_lin(((nt + 1) & 1) ? bstage_lds1 : bstage_lds0,
                             (unsigned long long)(uintptr_t)(W1t + (size_t)(nt + 1) * 16 * TWO_D),
                             QW_PANEL1);
                __builtin_amdgcn_s_wait_tensorcnt(1);  // TDM in-order: panel nt done
            } else {
                __builtin_amdgcn_s_wait_tensorcnt(0);
            }
        }
        __syncthreads();                              // panel nt visible to all waves

        const _Float16* Bpanel = bstage + (nt & 1) * 8192;  // [16 n][256 k] f16
        v8f acc = {0.f, 0.f, 0.f, 0.f, 0.f, 0.f, 0.f, 0.f};
        #pragma unroll
        for (int kt = 0; kt < TWO_D / 32; ++kt) {    // 8 k-steps
            v16h a = *(const v16h*)(featsW + m * TWO_D + kt * 32 + sub * 16);
            v16h b = *(const v16h*)(Bpanel + nl * TWO_D + kt * 32 + sub * 16);
            acc = __builtin_amdgcn_wmma_f32_16x16x32_f16(
                      false, a, false, b, (short)0, acc, false, false);
        }
        const float bias = cB1[nt * 16 + nl];
        #pragma unroll
        for (int r = 0; r < 8; ++r) {
            float v = acc[r] + bias;
            v = v > 0.f ? v : 0.f;                   // relu
            sum[r]   += v;
            sumsq[r] += v * v;
            const int mm = r + 8 * sub;              // C layout: M = r + 8*(lane/16)
            h1W[mm * HH + nt * 16 + nl] = (_Float16)v;
        }
        __syncthreads();                             // all done before buffer reuse
    }

    // ---- overlap: kick off GEMM2 panel 0 DMA while doing LayerNorm ----
    if (tid == 0) {
        tdm_load_lin(bstage_lds0, (unsigned long long)(uintptr_t)W2t, QW_PANEL2);
    }

    // ---- LayerNorm stats: half-wave tree reduction over 16 columns ----
    #pragma unroll
    for (int r = 0; r < 8; ++r) {
        float s = sum[r], q = sumsq[r];
        #pragma unroll
        for (int off = 1; off < 16; off <<= 1) {
            s += __shfl_xor(s, off, 32);
            q += __shfl_xor(q, off, 32);
        }
        if (nl == 0) {
            const int mm = r + 8 * sub;
            const float mean = s * (1.f / (float)HH);
            const float var  = q * (1.f / (float)HH) - mean * mean;
            statsW[2 * mm]     = mean;
            statsW[2 * mm + 1] = rsqrtf(var + EPSLN);
        }
    }

    // ---- apply LayerNorm in place on the h1 tile ----
    for (int idx = lane; idx < 16 * HH; idx += 32) {
        const int mm = idx >> 9;        // /512
        const int nn = idx & (HH - 1);
        float v = (float)h1W[idx];
        v = (v - statsW[2 * mm]) * statsW[2 * mm + 1] * cGamma[nn] + cBeta[nn];
        h1W[idx] = (_Float16)v;
    }

    // ---- GEMM2: h2 = relu(h1n @ W2 + b2), h2 tile reuses feats LDS ----
    _Float16* h2W = featsW;             // [16 x 256]
    #pragma unroll 1
    for (int nt = 0; nt < HH2 / 16; ++nt) {          // 16 n-tiles
        if (tid == 0) {
            if (nt + 1 < HH2 / 16) {
                tdm_load_lin(((nt + 1) & 1) ? bstage_lds1 : bstage_lds0,
                             (unsigned long long)(uintptr_t)(W2t + (size_t)(nt + 1) * 16 * HH),
                             QW_PANEL2);
                __builtin_amdgcn_s_wait_tensorcnt(1);
            } else {
                __builtin_amdgcn_s_wait_tensorcnt(0);
            }
        }
        __syncthreads();

        const _Float16* Bpanel = bstage + (nt & 1) * 8192;  // [16 n][512 k] f16
        v8f acc = {0.f, 0.f, 0.f, 0.f, 0.f, 0.f, 0.f, 0.f};
        #pragma unroll
        for (int kt = 0; kt < HH / 32; ++kt) {       // 16 k-steps
            v16h a = *(const v16h*)(h1W + m * HH + kt * 32 + sub * 16);
            v16h b = *(const v16h*)(Bpanel + nl * HH + kt * 32 + sub * 16);
            acc = __builtin_amdgcn_wmma_f32_16x16x32_f16(
                      false, a, false, b, (short)0, acc, false, false);
        }
        const float bias = cB2[nt * 16 + nl];
        #pragma unroll
        for (int r = 0; r < 8; ++r) {
            float v = acc[r] + bias;
            v = v > 0.f ? v : 0.f;
            h2W[(r + 8 * sub) * HH2 + nt * 16 + nl] = (_Float16)v;
        }
        __syncthreads();
    }

    // ---- GEMM3: logits = h2 @ W3 + b3 (VALU dot, one cross-half shuffle) ----
    {
        float s = 0.f;
        const _Float16* hrow = h2W + m * HH2 + sub * 128;
        const float* w3 = cW3 + sub * 128;
        #pragma unroll 4
        for (int k = 0; k < 128; ++k) s += (float)hrow[k] * w3[k];
        s += __shfl_xor(s, 16, 32);
        if (sub == 0) {
            const float b3v = b3[0];
            out[e] = s + b3v;
            // pairs output (reference returns int32; representable exactly as f32)
            out[EE + 2 * (size_t)e]     = (float)iIdx;
            out[EE + 2 * (size_t)e + 1] = (float)jIdx;
        }
    }
}

// ================= host-side launch ==========================================
extern "C" void kernel_launch(void* const* d_in, const int* in_sizes, int n_in,
                              void* d_out, int out_size, void* d_ws, size_t ws_size,
                              hipStream_t stream) {
    (void)in_sizes; (void)n_in; (void)out_size; (void)ws_size;

    const float* emb   = (const float*)d_in[0];
    const float* W1    = (const float*)d_in[1];
    const float* b1    = (const float*)d_in[2];
    const float* gamma = (const float*)d_in[3];
    const float* beta  = (const float*)d_in[4];
    const float* W2    = (const float*)d_in[5];
    const float* b2    = (const float*)d_in[6];
    const float* W3    = (const float*)d_in[7];
    const float* b3    = (const float*)d_in[8];

    _Float16* embH = (_Float16*)d_ws;
    _Float16* W1t  = embH + EMBH_ELEMS;
    _Float16* W2t  = W1t + W1T_ELEMS;

    // pack / transpose staging (393216 elements)
    {
        const int total = EMBH_ELEMS + W1T_ELEMS + W2T_ELEMS;
        edge_pack_kernel<<<(total + 255) / 256, 256, 0, stream>>>(
            emb, W1, W2, embH, W1t, W2t);
    }

    // fused MLP: 128 edges per block, E = 4092 * 128 exactly
    edge_mlp_kernel<<<EE / 128, 256, LDS_TOTAL, stream>>>(
        embH, W1t, W2t, b1, gamma, beta, b2, W3, b3, (float*)d_out);
}
